// GMN_embed_maxsim_dot_19335942766731
// MI455X (gfx1250) — matrix operand order repaired and checked by reading, not compiled, not run on hardware
//
#include <hip/hip_runtime.h>

// ---------------------------------------------------------------------------
// Problem constants (match reference)
// ---------------------------------------------------------------------------
#define NB     64            // pairs
#define NQ_    400
#define NC_    512
#define BLOCK_ 912           // NQ + NC
#define NN     (NB * BLOCK_) // 58368 nodes
#define DEG_   8
#define EE     (NN * DEG_)   // 466944 edges
#define HDIM   128
#define GDIM   256

// ---------------------------------------------------------------------------
// Types / helpers
// ---------------------------------------------------------------------------
typedef __bf16 bf16_t;
typedef bf16_t v16bf __attribute__((ext_vector_type(16)));
typedef float  v8f   __attribute__((ext_vector_type(8)));
typedef unsigned int v4u __attribute__((ext_vector_type(4)));
typedef int v4i __attribute__((ext_vector_type(4)));

union Frag16 { v16bf v; v4u u[2]; };
union Pack16 { unsigned short s[16]; v4u u[2]; };

// gfx1250 async-to-LDS path (ASYNCcnt), guarded so either toolchain compiles.
#if defined(__HIP_DEVICE_COMPILE__) && \
    __has_builtin(__builtin_amdgcn_global_load_async_to_lds_b128) && \
    __has_builtin(__builtin_amdgcn_s_wait_asynccnt)
#define USE_ASYNC 1
#else
#define USE_ASYNC 0
#endif

__device__ __forceinline__ void cp_b128(void* lds, const void* g) {
#if USE_ASYNC
    // param 0: v4i in addrspace(1) (printed as "__device__"), non-const
    // param 1: v4i in addrspace(3); then imm offset, imm cpol
    __builtin_amdgcn_global_load_async_to_lds_b128(
        (__attribute__((address_space(1))) v4i*)g,
        (__attribute__((address_space(3))) v4i*)lds, 0, 0);
#else
    *(v4u*)lds = *(const v4u*)g;
#endif
}
__device__ __forceinline__ void wait_async() {
#if USE_ASYNC
    __builtin_amdgcn_s_wait_asynccnt(0);
#endif
}

__device__ __forceinline__ unsigned short f2bf(float f) {
    unsigned u = __builtin_bit_cast(unsigned, f);
    unsigned r = u + 0x7FFFu + ((u >> 16) & 1u);   // round-to-nearest-even
    return (unsigned short)(r >> 16);
}
__device__ __forceinline__ float bf2f(unsigned short h) {
    unsigned u = ((unsigned)h) << 16;
    return __builtin_bit_cast(float, u);
}

static inline size_t align256(size_t x) { return (x + 255) & ~(size_t)255; }

// ---------------------------------------------------------------------------
// Small utility kernels
// ---------------------------------------------------------------------------
__global__ void k_zero_f32(float* __restrict__ p, long long n) {
    long long i = (long long)blockIdx.x * blockDim.x + threadIdx.x;
    if (i < n) p[i] = 0.0f;
}

// Pre-pack f32 weight W[K,N] into WMMA-B layout:
// Wpk[kp][n][p] (u32) = pack(bf16(W[kp*32+2p][n]), bf16(W[kp*32+2p+1][n]))
// so a 32xTN weight panel is a contiguous run of TN*16 dwords.
__global__ void k_prepack_w(const float* __restrict__ W,
                            unsigned int* __restrict__ Wpk,
                            int K, int N, int ndw) {
    int t = blockIdx.x * blockDim.x + threadIdx.x;
    if (t >= ndw) return;
    int p   = t & 15;
    int n   = (t >> 4) % N;
    int kp  = t / (N * 16);
    int gk  = kp * 32 + 2 * p;
    unsigned lo = (gk     < K) ? f2bf(W[(size_t)gk * N + n])       : 0u;
    unsigned hi = (gk + 1 < K) ? f2bf(W[(size_t)(gk + 1) * N + n]) : 0u;
    Wpk[t] = lo | (hi << 16);
}

// scatter-add: agg[to_idx[e]][c] += m[e][c]   (m is bf16, agg is f32)
__global__ void k_scatter(const unsigned short* __restrict__ m,
                          const int* __restrict__ to_idx,
                          float* __restrict__ agg) {
    long long t = (long long)blockIdx.x * blockDim.x + threadIdx.x;
    long long e = t >> 5;                 // 32 threads per edge
    if (e >= (long long)EE) return;
    int c0 = ((int)t & 31) * 4;
    int dst = to_idx[e];
    const unsigned short* mp = m + e * HDIM + c0;
    float* ap = agg + (size_t)dst * HDIM + c0;
    #pragma unroll
    for (int i = 0; i < 4; ++i) atomicAdd(ap + i, bf2f(mp[i]));
}

// gated aggregation: feats[r][c] = x[r][G+c] * sigmoid(x[r][c]), bf16 out
__global__ void k_gate(const float* __restrict__ x,
                       unsigned short* __restrict__ feats) {
    long long t = (long long)blockIdx.x * blockDim.x + threadIdx.x;
    if (t >= (long long)NN * GDIM) return;
    long long row = t >> 8;               // GDIM == 256
    int c = (int)t & 255;
    float g = x[row * (2 * GDIM) + c];
    float v = x[row * (2 * GDIM) + GDIM + c];
    float s = 1.0f / (1.0f + __expf(-g));
    feats[row * GDIM + c] = f2bf(v * s);
}

// ---------------------------------------------------------------------------
// Tiled WMMA GEMM:  C[M,N] = act( A[M,K] @ W[K,N] + bias ), W pre-packed.
//   MODE 0: A is f32 row-major
//   MODE 1: A is bf16 row-major
//   MODE 2: message gather: row r -> [ h[from[r]] | h[to[r]] | e[r] ]  (K=384)
//   MODE 3: update gather:  row r -> [ h[r] (bf16) | agg[r] (f32) ]    (K=256)
//
// Workgroup: 256 thr = 8 waves, tile 128x128, waves on 4(M) x 2(N);
// each wave owns a 32x64 slab: 2 A frags x 4 B frags -> 8 WMMAs per K-step.
// MODE 1/2 (pure bf16 sources): double-buffered LDS pipeline fed by
// async-to-LDS b128 copies (ASYNCcnt), staging the next K-chunk while the
// current one runs through the matrix pipe.
// MODE 0/3 (f32 conversion needed): register-pipelined single-buffer path.
// Fragments are pure ds_load_b128 (As row-major 96B stride; Wp pair-packed
// column-major). K multiple of 16, M/N multiples of 128 -> exact-run staging.
// ---------------------------------------------------------------------------
#define TM 128
#define TN 128
#define TK 32
#define AS_STRIDE 48   // ushorts; 96 B (multiple of 16 B)
#define WP_STRIDE 20   // dwords;  80 B (multiple of 16 B)

template <int MODE>
__global__ __launch_bounds__(256) void gemm_tile(
    const void* __restrict__ Aa,
    const unsigned int* __restrict__ Wpk,   // pre-packed weights
    const float* __restrict__ bias,
    void* __restrict__ Cc,
    int M, int K, int N,
    const unsigned short* __restrict__ hbuf,
    const unsigned short* __restrict__ ebuf,
    const float* __restrict__ aggbuf,
    const int* __restrict__ from_idx,
    const int* __restrict__ to_idx,
    int relu, int out_bf16)
{
    __shared__ __align__(16) unsigned short As[2][TM][AS_STRIDE];
    __shared__ __align__(16) unsigned int   Wp[2][TN][WP_STRIDE];

    const int tid  = threadIdx.x;
    const int lane = tid & 31;
    const int wave = tid >> 5;       // 0..7
    const int wm   = wave >> 1;      // 0..3 -> +32*wm rows
    const int wn   = wave & 1;       // 0..1 -> +64*wn cols
    const int tileM = blockIdx.x * TM;
    const int tileN = blockIdx.y * TN;

    const int lrow = lane & 15;
    const int kb   = (lane >> 4) * 16;  // A-frag elem offset within row
    const int q0   = (lane >> 4) * 8;   // B-frag dword offset within col

    // staging geometry (fixed per thread)
    const int ar  = tid >> 1;            // A row 0..127
    const int akq = (tid & 1) * 16;      // A k-offset 0/16
    const int wcn = tid >> 1;            // W col 0..127
    const int whalf = (tid & 1) * 8;     // W dword half 0/8
    const int agrow = tileM + ar;

    // per-row gather source pointers, resolved once
    const unsigned short* hrow_f = nullptr;
    const unsigned short* hrow_t = nullptr;
    const unsigned short* erow   = nullptr;
    if (MODE == 2) {
        hrow_f = hbuf + (size_t)from_idx[agrow] * HDIM;
        hrow_t = hbuf + (size_t)to_idx[agrow] * HDIM;
        erow   = ebuf + (size_t)agrow * HDIM;
    }

    v8f acc[2][4];
    #pragma unroll
    for (int i = 0; i < 2; ++i)
        #pragma unroll
        for (int s = 0; s < 4; ++s) acc[i][s] = (v8f){};

    auto do_wmma = [&](int buf) {
        v16bf afrag[2];
        #pragma unroll
        for (int i = 0; i < 2; ++i) {
            const unsigned short* rp = &As[buf][wm * 32 + i * 16 + lrow][0];
            Frag16 f;
            f.u[0] = *(const v4u*)(rp + kb);
            f.u[1] = *(const v4u*)(rp + kb + 8);
            afrag[i] = f.v;
        }
        v16bf bfrag[4];
        #pragma unroll
        for (int s = 0; s < 4; ++s) {
            const unsigned int* cp = &Wp[buf][wn * 64 + s * 16 + lrow][0];
            Frag16 f;
            f.u[0] = *(const v4u*)(cp + q0);
            f.u[1] = *(const v4u*)(cp + q0 + 4);
            bfrag[s] = f.v;
        }
        #pragma unroll
        for (int i = 0; i < 2; ++i)
            #pragma unroll
            for (int s = 0; s < 4; ++s)
                acc[i][s] = __builtin_amdgcn_wmma_f32_16x16x32_bf16(
                    false, afrag[i], false, bfrag[s], (short)0, acc[i][s],
                    false, false);
    };

    if constexpr (MODE == 1 || MODE == 2) {
        // ---- async double-buffered pipeline (pure bf16 sources) ----
        auto stage = [&](int k0, int buf) {
            const unsigned int* wsrc = Wpk +
                (((size_t)(k0 >> 5) * N + tileN + wcn) * 16 + whalf);
            cp_b128(&Wp[buf][wcn][whalf],     wsrc);
            cp_b128(&Wp[buf][wcn][whalf + 4], wsrc + 4);
            int gk0 = k0 + akq;   // always < K here (K multiple of 32)
            const unsigned short* src;
            if (MODE == 1) {
                src = (const unsigned short*)Aa + (size_t)agrow * K + gk0;
            } else {
                if (gk0 < HDIM)          src = hrow_f + gk0;
                else if (gk0 < 2 * HDIM) src = hrow_t + (gk0 - HDIM);
                else                     src = erow + (gk0 - 2 * HDIM);
            }
            cp_b128(&As[buf][ar][akq],     src);
            cp_b128(&As[buf][ar][akq + 8], src + 8);
        };
        stage(0, 0);
        const int nsteps = K / TK;
        for (int st = 0; st < nsteps; ++st) {
            int cur = st & 1;
            wait_async();
            __syncthreads();                       // buf[cur] ready for all waves
            if (st + 1 < nsteps) stage((st + 1) * TK, 1 - cur);
            do_wmma(cur);
            __syncthreads();                       // reads done before reuse
        }
    } else {
        // ---- register-pipelined path (needs f32 -> bf16 conversion) ----
        Pack16 areg;
        v4u    wreg0, wreg1;
        auto load_stage = [&](int k0) {
            const v4u* wsrc = (const v4u*)(Wpk +
                (((size_t)(k0 >> 5) * N + tileN + wcn) * 16 + whalf));
            wreg0 = wsrc[0];
            wreg1 = wsrc[1];
            int gk0 = k0 + akq;
            if (gk0 >= K) {
                areg.u[0] = (v4u){0, 0, 0, 0};
                areg.u[1] = (v4u){0, 0, 0, 0};
            } else if (MODE == 0) {
                const float* src = (const float*)Aa + (size_t)agrow * K + gk0;
                #pragma unroll
                for (int i = 0; i < 16; ++i) areg.s[i] = f2bf(src[i]);
            } else { // MODE 3
                if (gk0 < HDIM) {
                    const unsigned short* src =
                        hbuf + (size_t)agrow * HDIM + gk0;
                    areg.u[0] = ((const v4u*)src)[0];
                    areg.u[1] = ((const v4u*)src)[1];
                } else {
                    const float* src =
                        aggbuf + (size_t)agrow * HDIM + (gk0 - HDIM);
                    #pragma unroll
                    for (int i = 0; i < 16; ++i) areg.s[i] = f2bf(src[i]);
                }
            }
        };
        auto store_stage = [&]() {
            *(v4u*)&As[0][ar][akq]     = areg.u[0];
            *(v4u*)&As[0][ar][akq + 8] = areg.u[1];
            *(v4u*)&Wp[0][wcn][whalf]     = wreg0;
            *(v4u*)&Wp[0][wcn][whalf + 4] = wreg1;
        };
        load_stage(0);
        for (int k0 = 0; k0 < K; k0 += TK) {
            store_stage();
            __syncthreads();
            if (k0 + TK < K) load_stage(k0 + TK);   // overlap with WMMAs
            do_wmma(0);
            __syncthreads();
        }
    }

    // ---- epilogue: bias (+ReLU), store f32 or bf16 ----
    #pragma unroll
    for (int i = 0; i < 2; ++i) {
        int mbase = tileM + wm * 32 + i * 16 + (lane >> 4) * 8;
        #pragma unroll
        for (int s = 0; s < 4; ++s) {
            int col = tileN + wn * 64 + s * 16 + (lane & 15);
            float bv = bias[col];
            #pragma unroll
            for (int r = 0; r < 8; ++r) {
                float v = acc[i][s][r] + bv;
                if (relu) v = fmaxf(v, 0.0f);
                size_t off = (size_t)(mbase + r) * N + col;
                if (out_bf16) ((unsigned short*)Cc)[off] = f2bf(v);
                else          ((float*)Cc)[off] = v;
            }
        }
    }
}

// ---------------------------------------------------------------------------
// maxsim: per (pair, 16-query tile): scores = Q(16x256) @ C^T(256x512),
// running max over corpus, then sum of row-maxima atomically into out[b].
// 4 waves sweep 32 corpus tiles; fragments are direct global_load_b128
// (C^T layout is free: B(k,n)=c[n][k]).
// ---------------------------------------------------------------------------
#define QTILES 25   // 400 valid query rows = 25 * 16, padded rows contribute 0

__global__ __launch_bounds__(128) void k_maxsim(
    const unsigned short* __restrict__ feats, float* __restrict__ out)
{
    int b    = blockIdx.x / QTILES;
    int qt   = blockIdx.x % QTILES;
    int lane = threadIdx.x & 31;
    int wave = threadIdx.x >> 5;   // 0..3

    const unsigned short* qbase = feats + ((size_t)b * BLOCK_ + qt * 16) * GDIM;
    const unsigned short* cbase = feats + ((size_t)b * BLOCK_ + NQ_) * GDIM;

    // preload all 8 K-step A fragments for this query tile
    v16bf afrag[8];
    {
        int row = lane & 15;
        int kb  = (lane >> 4) * 16;
        const unsigned short* qr = qbase + (size_t)row * GDIM;
        #pragma unroll
        for (int ks = 0; ks < 8; ++ks) {
            Frag16 f;
            f.u[0] = *(const v4u*)(qr + ks * 32 + kb);
            f.u[1] = *(const v4u*)(qr + ks * 32 + kb + 8);
            afrag[ks] = f.v;
        }
    }

    float rmax[8];
    #pragma unroll
    for (int r = 0; r < 8; ++r) rmax[r] = -3.4e38f;

    for (int ct = wave; ct < NC_ / 16; ct += 4) {
        const unsigned short* cr =
            cbase + ((size_t)(ct * 16 + (lane & 15))) * GDIM;
        int kb2 = (lane >> 4) * 16;
        v8f acc = {};
        #pragma unroll
        for (int ks = 0; ks < 8; ++ks) {
            Frag16 f;
            f.u[0] = *(const v4u*)(cr + ks * 32 + kb2);
            f.u[1] = *(const v4u*)(cr + ks * 32 + kb2 + 8);
            acc = __builtin_amdgcn_wmma_f32_16x16x32_bf16(
                false, afrag[ks], false, f.v, (short)0, acc, false, false);
        }
        #pragma unroll
        for (int r = 0; r < 8; ++r) rmax[r] = fmaxf(rmax[r], acc[r]);
    }

    __shared__ float red[4][16][16];
    {
        int m0 = (lane >> 4) * 8;
        int n  = lane & 15;
        #pragma unroll
        for (int r = 0; r < 8; ++r) red[wave][m0 + r][n] = rmax[r];
    }
    __syncthreads();

    __shared__ float rowmax[16];
    if (threadIdx.x < 16) {
        float v = -3.4e38f;
        for (int w = 0; w < 4; ++w)
            for (int n = 0; n < 16; ++n)
                v = fmaxf(v, red[w][threadIdx.x][n]);
        rowmax[threadIdx.x] = v;
    }
    __syncthreads();
    if (threadIdx.x == 0) {
        float s = 0.0f;
        for (int i = 0; i < 16; ++i) s += rowmax[i];
        atomicAdd(&out[b], s);
    }
}

// ---------------------------------------------------------------------------
// Orchestration
// ---------------------------------------------------------------------------
extern "C" void kernel_launch(void* const* d_in, const int* in_sizes, int n_in,
                              void* d_out, int out_size, void* d_ws, size_t ws_size,
                              hipStream_t stream) {
    (void)in_sizes; (void)n_in; (void)out_size; (void)ws_size;

    const float* node_features = (const float*)d_in[0];
    const float* edge_features = (const float*)d_in[1];
    const int*   from_idx      = (const int*)d_in[2];
    const int*   to_idx        = (const int*)d_in[3];
    const float* enc_nw1 = (const float*)d_in[4];  const float* enc_nb1 = (const float*)d_in[5];
    const float* enc_nw2 = (const float*)d_in[6];  const float* enc_nb2 = (const float*)d_in[7];
    const float* enc_ew1 = (const float*)d_in[8];  const float* enc_eb1 = (const float*)d_in[9];
    const float* enc_ew2 = (const float*)d_in[10]; const float* enc_eb2 = (const float*)d_in[11];
    const float* msg_w1  = (const float*)d_in[12]; const float* msg_b1  = (const float*)d_in[13];
    const float* msg_w2  = (const float*)d_in[14]; const float* msg_b2  = (const float*)d_in[15];
    const float* upd_w1  = (const float*)d_in[16]; const float* upd_b1  = (const float*)d_in[17];
    const float* upd_w2  = (const float*)d_in[18]; const float* upd_b2  = (const float*)d_in[19];
    const float* agg_w   = (const float*)d_in[20]; const float* agg_b   = (const float*)d_in[21];
    float* out = (float*)d_out;

    // ---- workspace bump allocator ----
    char* ws = (char*)d_ws;
    size_t off = 0;
    auto alloc = [&](size_t bytes) -> char* {
        char* p = ws + off; off = align256(off + bytes); return p;
    };
    // pre-packed weights (u32 per bf16 pair): (K/32 panels) * N * 16 dwords
    auto wpkdw = [](int K, int N) { return (size_t)((K + 31) / 32) * N * 16; };
    unsigned int* p_nw1 = (unsigned int*)alloc(wpkdw(32,  128) * 4);
    unsigned int* p_nw2 = (unsigned int*)alloc(wpkdw(128, 128) * 4);
    unsigned int* p_ew1 = (unsigned int*)alloc(wpkdw(16,  128) * 4);
    unsigned int* p_ew2 = (unsigned int*)alloc(wpkdw(128, 128) * 4);
    unsigned int* p_mw1 = (unsigned int*)alloc(wpkdw(384, 256) * 4);
    unsigned int* p_mw2 = (unsigned int*)alloc(wpkdw(256, 128) * 4);
    unsigned int* p_uw1 = (unsigned int*)alloc(wpkdw(256, 256) * 4);
    unsigned int* p_uw2 = (unsigned int*)alloc(wpkdw(256, 128) * 4);
    unsigned int* p_ag  = (unsigned int*)alloc(wpkdw(128, 512) * 4);
    // activations
    unsigned short* h0   = (unsigned short*)alloc((size_t)NN * HDIM * 2);
    unsigned short* h1   = (unsigned short*)alloc((size_t)NN * HDIM * 2);
    unsigned short* ebuf = (unsigned short*)alloc((size_t)EE * HDIM * 2);
    unsigned short* mh   = (unsigned short*)alloc((size_t)EE * 2 * HDIM * 2); // msg hidden (also enc tmp)
    unsigned short* mbuf = (unsigned short*)alloc((size_t)EE * HDIM * 2);     // messages; aliased by x
    float*          agg  = (float*)alloc((size_t)NN * HDIM * 4);              // aliased by feats
    unsigned short* uh   = (unsigned short*)alloc((size_t)NN * 2 * HDIM * 2); // update hidden
    float*          x    = (float*)mbuf;                                      // [NN, 512] f32 (same bytes)
    unsigned short* feats = (unsigned short*)agg;                             // [NN, 256] bf16 (same bytes)
    unsigned short* tmp   = mh;                                               // encoder hidden scratch

    // ---- 1. pre-pack weights into WMMA-B layout ----
    auto prepack = [&](const float* s, unsigned int* d, int K, int N) {
        int ndw = (int)wpkdw(K, N);
        k_prepack_w<<<(ndw + 255) / 256, 256, 0, stream>>>(s, d, K, N, ndw);
    };
    prepack(enc_nw1, p_nw1, 32,  128);  prepack(enc_nw2, p_nw2, 128, 128);
    prepack(enc_ew1, p_ew1, 16,  128);  prepack(enc_ew2, p_ew2, 128, 128);
    prepack(msg_w1,  p_mw1, 384, 256);  prepack(msg_w2,  p_mw2, 256, 128);
    prepack(upd_w1,  p_uw1, 256, 256);  prepack(upd_w2,  p_uw2, 256, 128);
    prepack(agg_w,   p_ag,  128, 512);

    auto grid = [](int M, int N) { return dim3((unsigned)(M / TM), (unsigned)(N / TN)); };

    // ---- 2. encoders ----
    gemm_tile<0><<<grid(NN, 128), 256, 0, stream>>>(node_features, p_nw1, enc_nb1, tmp,
        NN, 32, 128, nullptr, nullptr, nullptr, nullptr, nullptr, 1, 1);
    gemm_tile<1><<<grid(NN, 128), 256, 0, stream>>>(tmp, p_nw2, enc_nb2, h0,
        NN, 128, 128, nullptr, nullptr, nullptr, nullptr, nullptr, 0, 1);
    gemm_tile<0><<<grid(EE, 128), 256, 0, stream>>>(edge_features, p_ew1, enc_eb1, tmp,
        EE, 16, 128, nullptr, nullptr, nullptr, nullptr, nullptr, 1, 1);
    gemm_tile<1><<<grid(EE, 128), 256, 0, stream>>>(tmp, p_ew2, enc_eb2, ebuf,
        EE, 128, 128, nullptr, nullptr, nullptr, nullptr, nullptr, 0, 1);

    // ---- 3. propagation layers (shared params) ----
    unsigned short* hc = h0;
    unsigned short* hn = h1;
    for (int it = 0; it < 3; ++it) {
        gemm_tile<2><<<grid(EE, 256), 256, 0, stream>>>(nullptr, p_mw1, msg_b1, mh,
            EE, 384, 256, hc, ebuf, nullptr, from_idx, to_idx, 1, 1);
        gemm_tile<1><<<grid(EE, 128), 256, 0, stream>>>(mh, p_mw2, msg_b2, mbuf,
            EE, 256, 128, nullptr, nullptr, nullptr, nullptr, nullptr, 0, 1);

        long long nAgg = (long long)NN * HDIM;
        k_zero_f32<<<(unsigned)((nAgg + 255) / 256), 256, 0, stream>>>(agg, nAgg);
        long long nScat = (long long)EE * 32;
        k_scatter<<<(unsigned)((nScat + 255) / 256), 256, 0, stream>>>(mbuf, to_idx, agg);

        gemm_tile<3><<<grid(NN, 256), 256, 0, stream>>>(nullptr, p_uw1, upd_b1, uh,
            NN, 256, 256, hc, nullptr, agg, nullptr, nullptr, 1, 1);
        gemm_tile<1><<<grid(NN, 128), 256, 0, stream>>>(uh, p_uw2, upd_b2, hn,
            NN, 256, 128, nullptr, nullptr, nullptr, nullptr, nullptr, 0, 1);
        unsigned short* t = hc; hc = hn; hn = t;
    }

    // ---- 4. aggregator + gating ----
    gemm_tile<1><<<grid(NN, 512), 256, 0, stream>>>(hc, p_ag, agg_b, x,
        NN, 128, 512, nullptr, nullptr, nullptr, nullptr, nullptr, 0, 0);
    long long nGate = (long long)NN * GDIM;
    k_gate<<<(unsigned)((nGate + 255) / 256), 256, 0, stream>>>(x, feats);

    // ---- 5. maxsim ----
    k_zero_f32<<<1, 64, 0, stream>>>(out, NB);
    k_maxsim<<<NB * QTILES, 128, 0, stream>>>(feats, out);
}
